// NodeModel_34239479283983
// MI455X (gfx1250) — matrix operand order repaired and checked by reading, not compiled, not run on hardware
//
#include <hip/hip_runtime.h>

typedef _Float16 v16h __attribute__((ext_vector_type(16)));
typedef _Float16 v8h  __attribute__((ext_vector_type(8)));
typedef _Float16 v4h  __attribute__((ext_vector_type(4)));
typedef float    v8f  __attribute__((ext_vector_type(8)));
typedef int      v4i  __attribute__((ext_vector_type(4)));

constexpr int N_NODES   = 200000;
constexpr int NODE_IN   = 128;
constexpr int HID       = 256;
constexpr int NODE_OUT  = 128;
constexpr int GLOBAL_IN = 64;
constexpr int CIN       = NODE_IN + GLOBAL_IN;   // 192

constexpr int M_TILE  = 128;                     // rows per block (8 waves x 16)
constexpr int THREADS = 256;                     // 8 wave32
constexpr int LSTRIDE = 264;                     // halves per activation LDS row (132 dw, 132%64=4 -> conflict-free)

// Packed-weight tiles: 32(K)x16(N) = 512 halves + 8-half skew per 8-lane group
// (lane offset = l*16 + (l>>3)*8) so the 32 b128 starts cover each 4-dword bank
// position exactly twice -> conflict-free ds_load_b128. Footprint 536 -> stride 544.
constexpr int TILE_HALVES = 544;
constexpr int W1_TILES = (CIN / 32) * (HID / 16);       // 96
constexpr int W2_TILES = (HID / 32) * (HID / 16);       // 128
constexpr int W3_TILES = (HID / 32) * (NODE_OUT / 16);  // 64
constexpr int W1_OFF = 0;
constexpr int W2_OFF = W1_TILES * TILE_HALVES;          // 52224
constexpr int W3_OFF = W2_OFF + W2_TILES * TILE_HALVES; // 121856
constexpr int PACK_LOGICAL = (W1_TILES + W2_TILES + W3_TILES) * 512;  // real elements

// LDS regions (halves): [ W (128 tiles) | H1 (128x264) | A/H2 (128x264) ]
constexpr int SMW_HALVES = W2_TILES * TILE_HALVES;      // 69632 (largest layer)
constexpr int BUF_HALVES = M_TILE * LSTRIDE;            // 33792
constexpr int SMEM_HALVES = SMW_HALVES + 2 * BUF_HALVES; // 137216 -> 268 KB

// ---------------------------------------------------------------------------
// Prologue: fp32 weights -> f16 B-fragment layout (ISA 16-bit B 32x16:
// k = 8*(lane>>4)+j for j<8, k = 16+8*(lane>>4)+(j-8) for j>=8).
// ---------------------------------------------------------------------------
__global__ void pack_weights_kernel(const float* __restrict__ W1,
                                    const float* __restrict__ W2,
                                    const float* __restrict__ W3,
                                    _Float16* __restrict__ packed) {
    int idx = blockIdx.x * blockDim.x + threadIdx.x;
    if (idx >= PACK_LOGICAL) return;
    const float* W; int ncols, rel, dbase;
    if (idx < W1_TILES * 512)                    { W = W1; ncols = HID;      rel = idx;                   dbase = W1_OFF; }
    else if (idx < (W1_TILES + W2_TILES) * 512)  { W = W2; ncols = HID;      rel = idx - W1_TILES * 512;  dbase = W2_OFF; }
    else                                         { W = W3; ncols = NODE_OUT; rel = idx - (W1_TILES + W2_TILES) * 512; dbase = W3_OFF; }
    int tile = rel >> 9;          // /512
    int lane = (rel >> 4) & 31;
    int j    = rel & 15;
    int ntiles = ncols >> 4;
    int kt = tile / ntiles, nt = tile % ntiles;
    int k  = kt * 32 + ((lane >> 4) << 3) + (j & 7) + ((j >> 3) << 4);
    int n  = nt * 16 + (lane & 15);
    int dst = dbase + tile * TILE_HALVES + lane * 16 + ((lane >> 3) << 3) + j;
    packed[dst] = (_Float16)W[k * ncols + n];
}

__device__ __forceinline__ v16h cat8(v8h lo, v8h hi) {
    return __builtin_shufflevector(lo, hi, 0, 1, 2, 3, 4, 5, 6, 7,
                                           8, 9, 10, 11, 12, 13, 14, 15);
}

// ---- async global->LDS staging (ASYNCcnt path), with safe fallback --------
__device__ __forceinline__ void copy16_async(const _Float16* g, _Float16* l) {
#if __has_builtin(__builtin_amdgcn_global_load_async_to_lds_b128)
    __builtin_amdgcn_global_load_async_to_lds_b128(
        (__attribute__((address_space(1))) v4i*)(__attribute__((address_space(1))) void*)(_Float16*)g,
        (__attribute__((address_space(3))) v4i*)(__attribute__((address_space(3))) void*)l,
        0, 0);
#else
    *(v8h*)l = *(const v8h*)g;
#endif
}
__device__ __forceinline__ void async_wait_all() {
#if __has_builtin(__builtin_amdgcn_s_wait_asynccnt)
    __builtin_amdgcn_s_wait_asynccnt(0);
#endif
}
__device__ __forceinline__ void stage_weights(const _Float16* __restrict__ gsrc,
                                              _Float16* __restrict__ ldst,
                                              int halves, int tid) {
    for (int i = tid * 8; i < halves; i += THREADS * 8)
        copy16_async(gsrc + i, ldst + i);
}

// ---- one MLP layer: LDS activations x LDS weights -> LDS (bias + ReLU) ----
template <int KT>
__device__ __forceinline__ void layer_lds(const _Float16* __restrict__ srcA,
                                          _Float16* __restrict__ dstH,
                                          const _Float16* __restrict__ smW,
                                          const float* __restrict__ bias,
                                          int rbase, int mrow, int kb,
                                          int ncol, int crow, int boff) {
    v16h af[KT];
    #pragma unroll
    for (int kt = 0; kt < KT; ++kt) {
        const _Float16* ap = srcA + (rbase + mrow) * LSTRIDE + kt * 32 + kb;
        af[kt] = cat8(*(const v8h*)ap, *(const v8h*)(ap + 16));
    }
    #pragma unroll
    for (int nt = 0; nt < 16; ++nt) {
        v8f acc = {0.f, 0.f, 0.f, 0.f, 0.f, 0.f, 0.f, 0.f};
        #pragma unroll
        for (int kt = 0; kt < KT; ++kt) {
            const _Float16* bp = smW + (kt * 16 + nt) * TILE_HALVES + boff;
            v16h bf = cat8(*(const v8h*)bp, *(const v8h*)(bp + 8));
            acc = __builtin_amdgcn_wmma_f32_16x16x32_f16(
                      false, af[kt], false, bf, (short)0, acc, false, false);
        }
        float b = bias[nt * 16 + ncol];
        #pragma unroll
        for (int v = 0; v < 8; ++v) {
            float r = fmaxf(acc[v] + b, 0.f);
            dstH[(rbase + crow + v) * LSTRIDE + nt * 16 + ncol] = (_Float16)r;
        }
    }
}

// ---------------------------------------------------------------------------
// Fused MLP: block = 128 rows; wave w owns row-tile w through all layers.
// Weights staged per-layer into LDS (async), shared by all 8 waves.
// ---------------------------------------------------------------------------
__global__ __launch_bounds__(THREADS)
void node_mlp_kernel(const float* __restrict__ x,
                     const float* __restrict__ u,
                     const int*   __restrict__ batch,
                     const float* __restrict__ b1,
                     const float* __restrict__ b2,
                     const float* __restrict__ b3,
                     const _Float16* __restrict__ packed,
                     float* __restrict__ out) {
    extern __shared__ _Float16 smem[];
    _Float16* smW   = smem;                            // per-layer weight tiles
    _Float16* smH1  = smem + SMW_HALVES;               // [128][264]
    _Float16* smAH2 = smem + SMW_HALVES + BUF_HALVES;  // staged A, then H2

    const int tid = threadIdx.x;
    const long long row0 = (long long)blockIdx.x * M_TILE;

    // ---- stage W1 (async) overlapped with A = [x | u[batch]] (f32->f16) ----
    stage_weights(packed + W1_OFF, smW, W1_TILES * TILE_HALVES, tid);
    for (int i = tid; i < M_TILE * (NODE_IN / 4); i += THREADS) {
        int r = i >> 5, c4 = i & 31;
        long long row = row0 + r; if (row >= N_NODES) row = N_NODES - 1;
        const float4 vv = *(const float4*)(x + row * NODE_IN + c4 * 4);
        v4h h = {(_Float16)vv.x, (_Float16)vv.y, (_Float16)vv.z, (_Float16)vv.w};
        *(v4h*)(smAH2 + r * LSTRIDE + c4 * 4) = h;
    }
    for (int i = tid; i < M_TILE * (GLOBAL_IN / 4); i += THREADS) {
        int r = i >> 4, c4 = i & 15;
        long long row = row0 + r; if (row >= N_NODES) row = N_NODES - 1;
        int bidx = batch[row];
        const float4 vv = *(const float4*)(u + (long long)bidx * GLOBAL_IN + c4 * 4);
        v4h h = {(_Float16)vv.x, (_Float16)vv.y, (_Float16)vv.z, (_Float16)vv.w};
        *(v4h*)(smAH2 + r * LSTRIDE + NODE_IN + c4 * 4) = h;
    }
    async_wait_all();
    __syncthreads();

    const int lane  = tid & 31;
    const int wv    = tid >> 5;             // wave id == row-tile id
    const int mrow  = lane & 15;            // A row within tile
    const int kb    = (lane >> 4) * 8;      // A K-offset for upper lane half
    const int ncol  = lane & 15;            // B/C column
    const int crow  = (lane >> 4) * 8;      // C row offset (M = v + crow)
    const int rbase = wv * 16;
    const int boff  = lane * 16 + ((lane >> 3) << 3);  // skewed B-frag offset

    // layer 1: [16x192] x [192x256] -> H1 (ReLU)
    layer_lds<6>(smAH2, smH1, smW, b1, rbase, mrow, kb, ncol, crow, boff);
    __syncthreads();                         // all waves done reading W1

    stage_weights(packed + W2_OFF, smW, W2_TILES * TILE_HALVES, tid);
    async_wait_all();
    __syncthreads();

    // layer 2: [16x256] x [256x256] -> H2 (ReLU), overwrites A region (own rows)
    layer_lds<8>(smH1, smAH2, smW, b2, rbase, mrow, kb, ncol, crow, boff);
    __syncthreads();                         // all waves done reading W2

    stage_weights(packed + W3_OFF, smW, W3_TILES * TILE_HALVES, tid);
    async_wait_all();
    __syncthreads();

    // layer 3: [16x256] x [256x128] + b3 + x residual -> out (fp32)
    {
        v16h af[8];
        #pragma unroll
        for (int kt = 0; kt < 8; ++kt) {
            const _Float16* ap = smAH2 + (rbase + mrow) * LSTRIDE + kt * 32 + kb;
            af[kt] = cat8(*(const v8h*)ap, *(const v8h*)(ap + 16));
        }
        #pragma unroll
        for (int nt = 0; nt < 8; ++nt) {
            v8f acc = {0.f, 0.f, 0.f, 0.f, 0.f, 0.f, 0.f, 0.f};
            #pragma unroll
            for (int kt = 0; kt < 8; ++kt) {
                const _Float16* bp = smW + (kt * 8 + nt) * TILE_HALVES + boff;
                v16h bf = cat8(*(const v8h*)bp, *(const v8h*)(bp + 8));
                acc = __builtin_amdgcn_wmma_f32_16x16x32_f16(
                          false, af[kt], false, bf, (short)0, acc, false, false);
            }
            float bias = b3[nt * 16 + ncol];
            #pragma unroll
            for (int v = 0; v < 8; ++v) {
                long long row = row0 + rbase + crow + v;
                if (row < N_NODES) {
                    int col = nt * 16 + ncol;
                    out[row * NODE_OUT + col] = acc[v] + bias + x[row * NODE_IN + col];
                }
            }
        }
    }
}

// ---------------------------------------------------------------------------
// Inputs (setup_inputs order): 0:x 1:edge_index(unused) 2:edge_attr(unused)
// 3:u 4:batch 5:W1 6:b1 7:W2 8:b2 9:W3 10:b3
// ---------------------------------------------------------------------------
extern "C" void kernel_launch(void* const* d_in, const int* in_sizes, int n_in,
                              void* d_out, int out_size, void* d_ws, size_t ws_size,
                              hipStream_t stream) {
    const float* x     = (const float*)d_in[0];
    const float* u     = (const float*)d_in[3];
    const int*   batch = (const int*)d_in[4];
    const float* W1    = (const float*)d_in[5];
    const float* b1    = (const float*)d_in[6];
    const float* W2    = (const float*)d_in[7];
    const float* b2    = (const float*)d_in[8];
    const float* W3    = (const float*)d_in[9];
    const float* b3    = (const float*)d_in[10];
    float* out = (float*)d_out;
    _Float16* packed = (_Float16*)d_ws;

    pack_weights_kernel<<<(PACK_LOGICAL + 255) / 256, 256, 0, stream>>>(W1, W2, W3, packed);

    size_t shmem = (size_t)SMEM_HALVES * sizeof(_Float16);  // 268 KB of 320 KB WGP LDS
    (void)hipFuncSetAttribute((const void*)node_mlp_kernel,
                              hipFuncAttributeMaxDynamicSharedMemorySize, (int)shmem);
    int grid = (N_NODES + M_TILE - 1) / M_TILE;  // 1563
    node_mlp_kernel<<<grid, THREADS, shmem, stream>>>(x, u, batch, b1, b2, b3, packed, out);
}
